// MultiHeadAttentionBlock_7756710936675
// MI455X (gfx1250) — matrix operand order repaired and checked by reading, not compiled, not run on hardware
//
#include <hip/hip_runtime.h>

#define S 2048
#define D 1024
#define H 16
#define DH 64

typedef __attribute__((ext_vector_type(16))) _Float16 v16h;
typedef __attribute__((ext_vector_type(8)))  _Float16 v8h;
typedef __attribute__((ext_vector_type(8)))  float    v8f;
typedef __attribute__((ext_vector_type(4)))  unsigned v4u;
typedef __attribute__((ext_vector_type(8)))  unsigned v8u;

union HF16 { v16h v; v8h h[2]; };

#define WMMA(a, b, c) __builtin_amdgcn_wmma_f32_16x16x32_f16(false, (a), false, (b), (short)0, (c), false, false)

// ---------------------------------------------------------------------------
// CDNA5 async global->LDS copy (16B per lane, tracked by ASYNCcnt)
// ---------------------------------------------------------------------------
__device__ __forceinline__ void async_b128(unsigned lds_off, const void* gaddr) {
  asm volatile("global_load_async_to_lds_b128 %0, %1, off"
               :: "v"(lds_off), "v"(gaddr) : "memory");
}

__device__ __forceinline__ void wait_async0() {
#if __has_builtin(__builtin_amdgcn_s_wait_asynccnt)
  __builtin_amdgcn_s_wait_asynccnt(0);
#else
  asm volatile("s_wait_asynccnt 0x0" ::: "memory");
#endif
}

__device__ __forceinline__ void wait_tensor0() {
#if __has_builtin(__builtin_amdgcn_s_wait_tensorcnt)
  __builtin_amdgcn_s_wait_tensorcnt(0);
#else
  asm volatile("s_wait_tensorcnt 0x0" ::: "memory");
#endif
}

// ---------------------------------------------------------------------------
// TDM: one instruction moves a 2D tile (tile_d1 rows x tile_d0 elems, f16)
// from global (row stride = stride_elems) into contiguous LDS at lds_addr.
// Descriptor per CDNA5 ISA (groups 0/1, 2D tensor).
// ---------------------------------------------------------------------------
__device__ __forceinline__ void tdm_load_2d(unsigned lds_addr, const void* gaddr,
                                            unsigned tile_d0, unsigned tile_d1,
                                            unsigned stride_elems) {
  const unsigned long long ga = (unsigned long long)gaddr;
  v4u g0;
  g0[0] = 1u;                                             // count=1 (valid D#)
  g0[1] = lds_addr;                                       // lds_addr [63:32]
  g0[2] = (unsigned)ga;                                   // global_addr lo
  g0[3] = ((unsigned)(ga >> 32) & 0x01FFFFFFu)            // global_addr hi
          | (2u << 30);                                   // type = 2 (image)
  v8u g1;
  g1[0] = 1u << 16;                                       // data_size=1 (2B)
  g1[1] = (tile_d0 & 0xFFFFu) << 16;                      // tensor_dim0 lo16
  g1[2] = (tile_d0 >> 16) | ((tile_d1 & 0xFFFFu) << 16);  // dim0 hi / dim1 lo
  g1[3] = (tile_d1 >> 16) | (tile_d0 << 16);              // dim1 hi / tile_dim0
  g1[4] = tile_d1;                                        // tile_dim1, tile_dim2=0
  g1[5] = stride_elems;                                   // tensor_dim0_stride lo32
  g1[6] = 0u;                                             // stride hi / dim1_stride lo
  g1[7] = 0u;
  asm volatile("tensor_load_to_lds %0, %1" :: "s"(g0), "s"(g1) : "memory");
}

// ---------------------------------------------------------------------------
// WMMA A-fragment loader from global (CDNA5 ISA §7.12.2, wave32)
// ---------------------------------------------------------------------------
__device__ __forceinline__ v16h load_a_frag(const _Float16* A, int ld, int m0, int k0) {
  const int lane = threadIdx.x & 31;
  const int row  = m0 + (lane & 15);
  const int kb   = k0 + ((lane >> 4) << 3);   // 0 or 8
  HF16 u;
  u.h[0] = *(const v8h*)(A + (size_t)row * ld + kb);
  u.h[1] = *(const v8h*)(A + (size_t)row * ld + kb + 16);
  return u.v;
}

// ---------------------------------------------------------------------------
// Kernel 1: per-row LayerNorm + cast to f16
// ---------------------------------------------------------------------------
__global__ __launch_bounds__(256) void ln_cast_kernel(
    const float* __restrict__ X, const float* __restrict__ G,
    const float* __restrict__ Bv, _Float16* __restrict__ XH) {
  const int s = blockIdx.x;
  const float* row = X + (size_t)s * D;
  float v[4]; float sum = 0.f, sq = 0.f;
  for (int i = 0; i < 4; ++i) {
    v[i] = row[threadIdx.x + i * 256];
    sum += v[i]; sq += v[i] * v[i];
  }
  for (int off = 16; off; off >>= 1) {
    sum += __shfl_xor(sum, off, 32);
    sq  += __shfl_xor(sq,  off, 32);
  }
  __shared__ float ssum[8], ssq[8];
  const int wave = threadIdx.x >> 5, lane = threadIdx.x & 31;
  if (lane == 0) { ssum[wave] = sum; ssq[wave] = sq; }
  __syncthreads();
  float ts = 0.f, tq = 0.f;
  for (int i = 0; i < 8; ++i) { ts += ssum[i]; tq += ssq[i]; }
  const float mu   = ts / (float)D;
  const float var  = tq / (float)D - mu * mu;
  const float rstd = rsqrtf(var + 1e-5f);
  for (int i = 0; i < 4; ++i) {
    const int d = threadIdx.x + i * 256;
    XH[(size_t)s * D + d] = (_Float16)((v[i] - mu) * rstd * G[d] + Bv[d]);
  }
}

// ---------------------------------------------------------------------------
// Kernel 2: elementwise f32 -> f16 cast (weights)
// ---------------------------------------------------------------------------
__global__ __launch_bounds__(256) void cast_f32_f16(const float* __restrict__ src,
                                                    _Float16* __restrict__ dst, int n) {
  const int i = blockIdx.x * 256 + threadIdx.x;
  if (i < n) dst[i] = (_Float16)src[i];
}

// ---------------------------------------------------------------------------
// Kernels 3/5: C[MxN] = A[MxK] * B[NxK]^T, double-buffered async->LDS staging.
// block = 256 (8 waves, 4x2), block tile 128x64, wave tile 32x32, K-step 32.
// ---------------------------------------------------------------------------
#define GEMM_MAINLOOP(A_, B_, ACC_)                                              \
  __shared__ _Float16 As[2][128 * 32];                                           \
  __shared__ _Float16 Bs[2][64 * 32];                                            \
  const int tid  = threadIdx.x;                                                  \
  const int wave = tid >> 5;                                                     \
  const int lane = tid & 31;                                                     \
  const int m0 = blockIdx.x * 128;                                               \
  const int n0 = blockIdx.y * 64;                                                \
  const int wm = (wave & 3) * 32;                                                \
  const int wn = (wave >> 2) * 32;                                               \
  const int crow = tid >> 2;            /* 0..63 */                              \
  const int cchk = (tid & 3) * 8;       /* half offset within 32-half row */     \
  const unsigned asBase = (unsigned)(size_t)(&As[0][0]);                         \
  const unsigned bsBase = (unsigned)(size_t)(&Bs[0][0]);                         \
  async_b128(asBase + (unsigned)((crow)      * 32 + cchk) * 2,                   \
             A_ + (size_t)(m0 + crow) * D + cchk);                               \
  async_b128(asBase + (unsigned)((crow + 64) * 32 + cchk) * 2,                   \
             A_ + (size_t)(m0 + crow + 64) * D + cchk);                          \
  async_b128(bsBase + (unsigned)((crow)      * 32 + cchk) * 2,                   \
             B_ + (size_t)(n0 + crow) * D + cchk);                               \
  for (int k0 = 0; k0 < D; k0 += 32) {                                           \
    const int cur = (k0 >> 5) & 1;                                               \
    wait_async0();                                                               \
    __syncthreads();                                                             \
    if (k0 + 32 < D) {                                                           \
      const unsigned nbA = (unsigned)((cur ^ 1) * 128 * 32);                     \
      const unsigned nbB = (unsigned)((cur ^ 1) * 64 * 32);                      \
      const int kk = k0 + 32;                                                    \
      async_b128(asBase + (nbA + (crow)      * 32 + cchk) * 2,                   \
                 A_ + (size_t)(m0 + crow) * D + kk + cchk);                      \
      async_b128(asBase + (nbA + (crow + 64) * 32 + cchk) * 2,                   \
                 A_ + (size_t)(m0 + crow + 64) * D + kk + cchk);                 \
      async_b128(bsBase + (nbB + (crow)      * 32 + cchk) * 2,                   \
                 B_ + (size_t)(n0 + crow) * D + kk + cchk);                      \
    }                                                                            \
    const _Float16* at = &As[cur][0];                                            \
    const _Float16* bt = &Bs[cur][0];                                            \
    const int arow = wm + (lane & 15);                                           \
    const int akb  = (lane >> 4) << 3;                                           \
    const int brow = wn + (lane & 15);                                           \
    const int bkb  = (lane >> 4) << 3;                                           \
    HF16 a0, a1, b0, b1;                                                         \
    a0.h[0] = *(const v8h*)(at + (arow)      * 32 + akb);                        \
    a0.h[1] = *(const v8h*)(at + (arow)      * 32 + akb + 16);                   \
    a1.h[0] = *(const v8h*)(at + (arow + 16) * 32 + akb);                        \
    a1.h[1] = *(const v8h*)(at + (arow + 16) * 32 + akb + 16);                   \
    b0.h[0] = *(const v8h*)(bt + (brow)      * 32 + bkb);                        \
    b0.h[1] = *(const v8h*)(bt + (brow)      * 32 + bkb + 16);                   \
    b1.h[0] = *(const v8h*)(bt + (brow + 16) * 32 + bkb);                        \
    b1.h[1] = *(const v8h*)(bt + (brow + 16) * 32 + bkb + 16);                   \
    ACC_[0][0] = WMMA(a0.v, b0.v, ACC_[0][0]);                                   \
    ACC_[0][1] = WMMA(a0.v, b1.v, ACC_[0][1]);                                   \
    ACC_[1][0] = WMMA(a1.v, b0.v, ACC_[1][0]);                                   \
    ACC_[1][1] = WMMA(a1.v, b1.v, ACC_[1][1]);                                   \
  }

__global__ __launch_bounds__(256) void gemm_qkv_kernel(
    const _Float16* __restrict__ A, const _Float16* __restrict__ B,
    _Float16* __restrict__ Crm, _Float16* __restrict__ Ctr, int trans) {
  v8f acc[2][2] = {};
  GEMM_MAINLOOP(A, B, acc)
  const int hi = lane >> 4, colb = lane & 15;
  for (int mt = 0; mt < 2; ++mt)
    for (int nt = 0; nt < 2; ++nt) {
      const int n  = n0 + wn + nt * 16 + colb;
      const int mb = m0 + wm + mt * 16 + hi * 8;
      if (!trans) {
        for (int r = 0; r < 8; ++r)
          Crm[(size_t)(mb + r) * D + n] = (_Float16)acc[mt][nt][r];
      } else {
        v8h t;
        for (int r = 0; r < 8; ++r) t[r] = (_Float16)acc[mt][nt][r];
        *(v8h*)(Ctr + (size_t)n * S + mb) = t;   // contiguous in m
      }
    }
}

__global__ __launch_bounds__(256) void out_proj_kernel(
    const _Float16* __restrict__ A, const _Float16* __restrict__ B,
    const float* __restrict__ resid, float* __restrict__ Out) {
  v8f acc[2][2] = {};
  GEMM_MAINLOOP(A, B, acc)
  const int hi = lane >> 4, colb = lane & 15;
  for (int mt = 0; mt < 2; ++mt)
    for (int nt = 0; nt < 2; ++nt) {
      const int n  = n0 + wn + nt * 16 + colb;
      const int mb = m0 + wm + mt * 16 + hi * 8;
      for (int r = 0; r < 8; ++r) {
        const size_t idx = (size_t)(mb + r) * D + n;
        Out[idx] = acc[mt][nt][r] + resid[idx];
      }
    }
}

// ---------------------------------------------------------------------------
// Kernel 4: flash attention. Block = 4 waves sharing one head; each wave owns
// a 16-query tile. K/V tiles staged in LDS by the Tensor Data Mover (one
// tensor_load_to_lds per tile, issued by wave 0, double buffered, TENSORcnt).
// Q,K row-major [S][D]; VT transposed [H*DH][S]; CTX row-major [S][D] f16.
// ---------------------------------------------------------------------------
__global__ __launch_bounds__(128) void attn_kernel(
    const _Float16* __restrict__ Q, const _Float16* __restrict__ K,
    const _Float16* __restrict__ VT, _Float16* __restrict__ CTX) {
  __shared__ _Float16 Ks[2][32 * 64];   // [key][feature]
  __shared__ _Float16 Vs[2][64 * 32];   // [dh][key]
  __shared__ _Float16 pbuf[4][16 * 32];
  const int wave = threadIdx.x >> 5;
  const int lane = threadIdx.x & 31;
  const int h  = blockIdx.y;
  const int m0 = blockIdx.x * 64 + wave * 16;
  const int hi = lane >> 4, colb = lane & 15;
  _Float16* pt = pbuf[wave];

  const _Float16* Qh  = Q + h * DH;
  const _Float16* Kh  = K + h * DH;
  const _Float16* VTh = VT + (size_t)(h * DH) * S;

  const v16h qa0 = load_a_frag(Qh, D, m0, 0);
  const v16h qa1 = load_a_frag(Qh, D, m0, 32);

  float mrun[8], lrun[8];
  for (int r = 0; r < 8; ++r) { mrun[r] = -3.0e38f; lrun[r] = 0.f; }
  v8f acc[4] = {};

  if (wave == 0) {
    tdm_load_2d((unsigned)(size_t)&Ks[0][0], Kh,  64, 32, D);
    tdm_load_2d((unsigned)(size_t)&Vs[0][0], VTh, 32, 64, S);
  }

  for (int kc = 0; kc < S; kc += 32) {
    const int cur = (kc >> 5) & 1;
    if (wave == 0) wait_tensor0();
    __syncthreads();                       // tiles[cur] visible to all waves
    if (wave == 0 && kc + 32 < S) {        // prefetch next chunk via TDM
      tdm_load_2d((unsigned)(size_t)&Ks[cur ^ 1][0], Kh + (size_t)(kc + 32) * D, 64, 32, D);
      tdm_load_2d((unsigned)(size_t)&Vs[cur ^ 1][0], VTh + (kc + 32),            32, 64, S);
    }
    const _Float16* kt  = &Ks[cur][0];
    const _Float16* vtl = &Vs[cur][0];

    // scores: 16 queries x 32 keys, contraction over DH=64 (2 wmma per chunk)
    HF16 kf[2][2];
    for (int c = 0; c < 2; ++c)
      for (int f = 0; f < 2; ++f) {
        const int col = c * 16 + colb;
        const int kb  = f * 32 + (hi << 3);
        kf[c][f].h[0] = *(const v8h*)(kt + col * 64 + kb);
        kf[c][f].h[1] = *(const v8h*)(kt + col * 64 + kb + 16);
      }
    v8f s0 = {}, s1 = {};
    s0 = WMMA(qa0, kf[0][0].v, s0); s0 = WMMA(qa1, kf[0][1].v, s0);
    s1 = WMMA(qa0, kf[1][0].v, s1); s1 = WMMA(qa1, kf[1][1].v, s1);

    // online softmax in f32 on the C layout: row = r + 8*hi, col = colb (+16)
    float p0[8], p1[8], mx[8];
    for (int r = 0; r < 8; ++r) {
      p0[r] = s0[r] * 0.125f;             // 1/sqrt(64)
      p1[r] = s1[r] * 0.125f;
      mx[r] = fmaxf(p0[r], p1[r]);
    }
    for (int off = 8; off; off >>= 1)
      for (int r = 0; r < 8; ++r)
        mx[r] = fmaxf(mx[r], __shfl_xor(mx[r], off, 16));
    float alpha[8];
    for (int r = 0; r < 8; ++r) {
      const float nm = fmaxf(mrun[r], mx[r]);
      alpha[r] = __expf(mrun[r] - nm);
      mrun[r]  = nm;
    }
    float rs[8];
    for (int r = 0; r < 8; ++r) {
      p0[r] = __expf(p0[r] - mrun[r]);
      p1[r] = __expf(p1[r] - mrun[r]);
      rs[r] = p0[r] + p1[r];
    }
    for (int off = 8; off; off >>= 1)
      for (int r = 0; r < 8; ++r)
        rs[r] += __shfl_xor(rs[r], off, 16);
    for (int r = 0; r < 8; ++r) lrun[r] = lrun[r] * alpha[r] + rs[r];
    for (int t = 0; t < 4; ++t)
      for (int r = 0; r < 8; ++r) acc[t][r] *= alpha[r];

    // P: C layout -> A layout through per-wave LDS tile (in-order per wave)
    for (int r = 0; r < 8; ++r) {
      pt[(r + 8 * hi) * 32 + colb]      = (_Float16)p0[r];
      pt[(r + 8 * hi) * 32 + 16 + colb] = (_Float16)p1[r];
    }
    asm volatile("s_wait_dscnt 0x0" ::: "memory");
    HF16 pa;
    const int prow = lane & 15, pkb = hi * 8;
    pa.h[0] = *(const v8h*)(pt + prow * 32 + pkb);
    pa.h[1] = *(const v8h*)(pt + prow * 32 + 16 + pkb);

    // ctx += P(16x32) x V(32x64): V B-frags contiguous in the staged Vs tile
    for (int nt = 0; nt < 4; ++nt) {
      HF16 u;
      const int row = nt * 16 + colb;
      const int sb  = hi << 3;
      u.h[0] = *(const v8h*)(vtl + row * 32 + sb);
      u.h[1] = *(const v8h*)(vtl + row * 32 + sb + 16);
      acc[nt] = WMMA(pa.v, u.v, acc[nt]);
    }
  }

  float inv[8];
  for (int r = 0; r < 8; ++r) inv[r] = 1.0f / lrun[r];
  for (int nt = 0; nt < 4; ++nt)
    for (int r = 0; r < 8; ++r)
      CTX[(size_t)(m0 + r + 8 * hi) * D + h * DH + nt * 16 + colb] =
          (_Float16)(acc[nt][r] * inv[r]);
}

// ---------------------------------------------------------------------------
// Launch
// ---------------------------------------------------------------------------
extern "C" void kernel_launch(void* const* d_in, const int* in_sizes, int n_in,
                              void* d_out, int out_size, void* d_ws, size_t ws_size,
                              hipStream_t stream) {
  (void)in_sizes; (void)n_in; (void)out_size; (void)ws_size;
  const float* x   = (const float*)d_in[0];
  const float* lnw = (const float*)d_in[1];
  const float* lnb = (const float*)d_in[2];
  const float* Wq  = (const float*)d_in[3];
  const float* Wk  = (const float*)d_in[4];
  const float* Wv  = (const float*)d_in[5];
  const float* Wo  = (const float*)d_in[6];
  char* ws = (char*)d_ws;

  _Float16* xh  = (_Float16*)(ws + ((size_t) 0 << 20));  // [S][D]   4 MB
  _Float16* wqh = (_Float16*)(ws + ((size_t) 4 << 20));  // [D][D]   2 MB
  _Float16* wkh = (_Float16*)(ws + ((size_t) 6 << 20));
  _Float16* wvh = (_Float16*)(ws + ((size_t) 8 << 20));
  _Float16* woh = (_Float16*)(ws + ((size_t)10 << 20));
  _Float16* qh  = (_Float16*)(ws + ((size_t)12 << 20));  // [S][D]
  _Float16* kh  = (_Float16*)(ws + ((size_t)16 << 20));  // [S][D]
  _Float16* vth = (_Float16*)(ws + ((size_t)20 << 20));  // [D][S] transposed V
  _Float16* cth = (_Float16*)(ws + ((size_t)24 << 20));  // [S][D] ctx

  const int NW = D * D;
  cast_f32_f16<<<NW / 256, 256, 0, stream>>>(Wq, wqh, NW);
  cast_f32_f16<<<NW / 256, 256, 0, stream>>>(Wk, wkh, NW);
  cast_f32_f16<<<NW / 256, 256, 0, stream>>>(Wv, wvh, NW);
  cast_f32_f16<<<NW / 256, 256, 0, stream>>>(Wo, woh, NW);

  ln_cast_kernel<<<S, 256, 0, stream>>>(x, lnw, lnb, xh);

  dim3 gg(S / 128, D / 64);
  gemm_qkv_kernel<<<gg, 256, 0, stream>>>(xh, wqh, qh,  nullptr, 0);
  gemm_qkv_kernel<<<gg, 256, 0, stream>>>(xh, wkh, kh,  nullptr, 0);
  gemm_qkv_kernel<<<gg, 256, 0, stream>>>(xh, wvh, nullptr, vth, 1);

  attn_kernel<<<dim3(S / 64, H), 128, 0, stream>>>(qh, kh, vth, cth);

  out_proj_kernel<<<dim3(S / 128, D / 64), 256, 0, stream>>>(cth, woh, x, (float*)d_out);
}
